// SingleHeadAttention_7911329759384
// MI455X (gfx1250) — matrix operand order repaired and checked by reading, not compiled, not run on hardware
//
#include <hip/hip_runtime.h>
#include <hip/hip_bf16.h>
#include <math.h>

// Problem constants (match reference): B=4, S=4096, H=2048, D=128
#define B_SZ   4
#define S_LEN  4096
#define H_DIM  2048
#define D_HEAD 128

typedef __attribute__((ext_vector_type(16))) __bf16 v16bf;
typedef __attribute__((ext_vector_type(8)))  float  v8f;

union FragBF16 {
    v16bf  v;
    uint4  u[2];
    __bf16 h[16];
};

static __device__ __forceinline__ v8f wmma_bf16(v16bf a, v16bf b, v8f c) {
    // D(16x16,f32) = A(16x32,bf16) x B(32x16,bf16) + C
    return __builtin_amdgcn_wmma_f32_16x16x32_bf16(
        /*neg_a=*/false, a, /*neg_b=*/false, b,
        /*c_mod=*/(short)0, c, /*reuse_a=*/false, /*reuse_b=*/false);
}

// -----------------------------------------------------------------------------
// Kernel 1 (fused, double-buffered): q,k,v = x @ {Wq,Wk,Wv} + {bq,bk,bv}
// Block: 384 threads = 12 waves. Waves 0-3 -> q, 4-7 -> k, 8-11 -> v.
// The 64-row x tile is staged into LDS ONCE per k-step and shared by all three
// matrices (x = 134 MB is the dominant HBM stream; read exactly once).
// Tile k0+32 is staged while tile k0 is computed (one barrier per iteration).
// -----------------------------------------------------------------------------
__global__ __launch_bounds__(384)
void qkv_proj_fused(const float* __restrict__ x,
                    const float* __restrict__ Wq, const float* __restrict__ bq,
                    const float* __restrict__ Wk, const float* __restrict__ bk,
                    const float* __restrict__ Wv, const float* __restrict__ bv,
                    unsigned short* __restrict__ q_out,
                    unsigned short* __restrict__ k_out,
                    unsigned short* __restrict__ v_out)
{
    // Stride 40 pad: B-frag reads are 16B-aligned ds_load_b128, conflict-free.
    __shared__ __align__(16) __bf16 ldsWT[2][3][D_HEAD][40];  // 61440 B
    __shared__ __align__(16) __bf16 ldsX[2][64][40];          // 10240 B

    const int tid    = threadIdx.x;
    const int wave   = tid >> 5;          // 0..11
    const int lane   = tid & 31;
    const int lane15 = lane & 15;
    const int hi     = lane >> 4;
    const int matId  = wave >> 2;         // 0:q 1:k 2:v   (== tid >> 7)
    const int wInMat = wave & 3;          // 16-row slice within the 64-row tile

    const float* W;
    const float* bias;
    unsigned short* outp;
    if (matId == 0)      { W = Wq; bias = bq; outp = q_out; }
    else if (matId == 1) { W = Wk; bias = bk; outp = k_out; }
    else                 { W = Wv; bias = bv; outp = v_out; }

    const int rowBase = blockIdx.x * 64;

    // ---- staging: W chunk (transposed, bf16) + x tile (row-major, bf16) ----
    auto stage = [&](int buf, int k0) {
        {   // this matrix's W[k0:k0+32, 0:128] -> ldsWT[buf][matId] (128 thr/mat)
            const int col = tid & 127;
            #pragma unroll 8
            for (int r = 0; r < 32; ++r)
                ldsWT[buf][matId][col][r] = (__bf16)W[(size_t)(k0 + r) * D_HEAD + col];
        }
        if (tid < 256) {  // x tile 64 rows x 32 k, fp32 -> bf16
            const int row = tid >> 2;
            const int c0  = (tid & 3) * 8;
            const float* bp = x + (size_t)(rowBase + row) * H_DIM + k0 + c0;
            const float4 f0 = ((const float4*)bp)[0];
            const float4 f1 = ((const float4*)bp)[1];
            FragBF16 h8;
            h8.h[0]=(__bf16)f0.x; h8.h[1]=(__bf16)f0.y; h8.h[2]=(__bf16)f0.z; h8.h[3]=(__bf16)f0.w;
            h8.h[4]=(__bf16)f1.x; h8.h[5]=(__bf16)f1.y; h8.h[6]=(__bf16)f1.z; h8.h[7]=(__bf16)f1.w;
            *(uint4*)&ldsX[buf][row][c0] = h8.u[0];
            if (k0 + 32 < H_DIM) __builtin_prefetch(bp + 32, 0, 0);  // next x chunk (HBM)
        }
    };

    v8f acc[8];
    #pragma unroll
    for (int i = 0; i < 8; ++i) acc[i] = (v8f){};

    stage(0, 0);
    __syncthreads();

    int buf = 0;
    for (int k0 = 0; k0 < H_DIM; k0 += 32, buf ^= 1) {
        if (k0 + 32 < H_DIM) stage(buf ^ 1, k0 + 32);   // overlap with compute

        // A fragment (contiguous k order, matches B):
        // lane<16: K={0..15}; lane>=16: K={16..31}
        FragBF16 a;
        {
            const int row = wInMat * 16 + lane15;
            a.u[0] = ((const uint4*)&ldsX[buf][row][hi * 16])[0];
            a.u[1] = ((const uint4*)&ldsX[buf][row][hi * 16 + 8])[0];
        }

        // Ping-pong B-fragment registers so frag nt+1's loads are in flight
        // while WMMA nt executes (wait dscnt<=2 instead of 0).
        FragBF16 f0, f1;
        auto loadB = [&](FragBF16& f, int nt) {
            const uint4* p = (const uint4*)&ldsWT[buf][matId][nt * 16 + lane15][hi * 16];
            f.u[0] = p[0];
            f.u[1] = p[1];
        };
        loadB(f0, 0);
        #pragma unroll
        for (int nt = 0; nt < 8; ++nt) {
            if (nt < 7) loadB((nt & 1) ? f0 : f1, nt + 1);
            acc[nt] = wmma_bf16(a.v, ((nt & 1) ? f1 : f0).v, acc[nt]);
        }
        __syncthreads();   // staging of buf^1 complete; buf free for reuse
    }

    // bias + store bf16 (C layout: VGPR j -> row j + 8*hi, col across lanes)
    #pragma unroll
    for (int nt = 0; nt < 8; ++nt) {
        const int col = nt * 16 + lane15;
        const float bb = bias[col];
        #pragma unroll
        for (int j = 0; j < 8; ++j) {
            const int row = rowBase + wInMat * 16 + j + hi * 8;
            const __bf16 h = (__bf16)(acc[nt][j] + bb);
            outp[(size_t)row * D_HEAD + col] = *(const unsigned short*)&h;
        }
    }
}

// -----------------------------------------------------------------------------
// Kernel 2: causal flash attention with post-softmax 1/sqrt(D) scaling.
// Block: 128 threads (4 waves), 64 query rows/block, key tiles of 32.
// V tile kt+1 is staged (transposed) while tile kt is computed.
// -----------------------------------------------------------------------------
__global__ __launch_bounds__(128)
void flash_attn_kernel(const unsigned short* __restrict__ qg,
                       const unsigned short* __restrict__ kg,
                       const unsigned short* __restrict__ vg,
                       float* __restrict__ out)
{
    __shared__ __align__(16) __bf16 ldsVT[2][D_HEAD][40];  // V^T tiles (double buf)
    __shared__ __align__(16) __bf16 ldsP[4][16][40];       // per-wave P relayout

    const int tid    = threadIdx.x;
    const int wave   = tid >> 5;
    const int lane   = tid & 31;
    const int lane15 = lane & 15;
    const int hi     = lane >> 4;

    const int b    = blockIdx.y;
    const int q0   = blockIdx.x * 64;
    const int qRow = q0 + wave * 16;
    const size_t seqBase = (size_t)b * S_LEN;

    const int ktEnd = blockIdx.x * 2 + 2;   // key tiles of 32 covering keys <= q0+63

    auto stageV = [&](int buf, int kbase) {
        const int key = tid & 31;
        const int d0  = (tid >> 5) * 32;
        union { uint4 u[4]; __bf16 h[32]; } vr;
        const __bf16* vbase = (const __bf16*)vg + (seqBase + kbase + key) * D_HEAD + d0;
        #pragma unroll
        for (int i = 0; i < 4; ++i) vr.u[i] = ((const uint4*)vbase)[i];
        #pragma unroll
        for (int i = 0; i < 32; ++i) ldsVT[buf][d0 + i][key] = vr.h[i];
        if (kbase + 32 < ktEnd * 32)          // prefetch the tile after next
            __builtin_prefetch(vbase + 32 * D_HEAD, 0, 0);
    };

    // Preload Q A-fragments for this wave's 16 rows (4 k-steps of 32 dims).
    FragBF16 aq[4];
    {
        const __bf16* qrow = (const __bf16*)qg + (seqBase + qRow + lane15) * D_HEAD;
        #pragma unroll
        for (int ks = 0; ks < 4; ++ks) {
            aq[ks].u[0] = ((const uint4*)(qrow + ks * 32 + hi * 16))[0];
            aq[ks].u[1] = ((const uint4*)(qrow + ks * 32 + hi * 16 + 8))[0];
        }
    }

    v8f o[8];
    #pragma unroll
    for (int i = 0; i < 8; ++i) o[i] = (v8f){};
    float m[8], l[8];
    #pragma unroll
    for (int j = 0; j < 8; ++j) { m[j] = -INFINITY; l[j] = 0.0f; }

    stageV(0, 0);
    __syncthreads();

    int buf = 0;
    for (int kt = 0; kt < ktEnd; ++kt, buf ^= 1) {
        const int kbase = kt * 32;
        if (kt + 1 < ktEnd) stageV(buf ^ 1, kbase + 32);   // overlap with compute

        if (kbase <= qRow + 15) {   // wave-uniform causal skip
            // ---- scores: batch all 8 K B-fragments (one VMEM clause) then WMMAs
            const __bf16* krow0 = (const __bf16*)kg + (seqBase + kbase + lane15) * D_HEAD;
            const __bf16* krow1 = krow0 + 16 * D_HEAD;
            FragBF16 bk[8];   // [2*ks] -> s0, [2*ks+1] -> s1
            #pragma unroll
            for (int ks = 0; ks < 4; ++ks) {
                const uint4* p0 = (const uint4*)(krow0 + ks * 32 + hi * 16);
                bk[2 * ks].u[0] = p0[0]; bk[2 * ks].u[1] = p0[1];
                const uint4* p1 = (const uint4*)(krow1 + ks * 32 + hi * 16);
                bk[2 * ks + 1].u[0] = p1[0]; bk[2 * ks + 1].u[1] = p1[1];
            }
            if (kt + 1 < ktEnd)             // prefetch next K tile
                __builtin_prefetch(krow0 + 32 * D_HEAD, 0, 0);
            v8f s0 = (v8f){}, s1 = (v8f){};
            #pragma unroll
            for (int ks = 0; ks < 4; ++ks) {
                s0 = wmma_bf16(aq[ks].v, bk[2 * ks].v, s0);
                s1 = wmma_bf16(aq[ks].v, bk[2 * ks + 1].v, s1);
            }

            // ---- causal mask + online softmax (fp32)
            float p0v[8], p1v[8];
            #pragma unroll
            for (int j = 0; j < 8; ++j) {
                const int qi  = qRow + j + hi * 8;
                const int ki  = kbase + lane15;
                float v0 = (ki      <= qi) ? s0[j] : -INFINITY;
                float v1 = (ki + 16 <= qi) ? s1[j] : -INFINITY;
                float t = fmaxf(v0, v1);
                t = fmaxf(t, __shfl_xor(t, 1, 32));
                t = fmaxf(t, __shfl_xor(t, 2, 32));
                t = fmaxf(t, __shfl_xor(t, 4, 32));
                t = fmaxf(t, __shfl_xor(t, 8, 32));
                const float mn   = fmaxf(m[j], t);        // finite after first tile
                const float corr = __expf(m[j] - mn);
                const float e0 = __expf(v0 - mn);
                const float e1 = __expf(v1 - mn);
                float rs = e0 + e1;
                rs += __shfl_xor(rs, 1, 32);
                rs += __shfl_xor(rs, 2, 32);
                rs += __shfl_xor(rs, 4, 32);
                rs += __shfl_xor(rs, 8, 32);
                l[j] = l[j] * corr + rs;
                m[j] = mn;
                p0v[j] = e0; p1v[j] = e1;
                #pragma unroll
                for (int nt = 0; nt < 8; ++nt) o[nt][j] *= corr;
            }

            // ---- P (C layout) -> LDS -> A layout bf16
            #pragma unroll
            for (int j = 0; j < 8; ++j) {
                ldsP[wave][j + hi * 8][lane15]      = (__bf16)p0v[j];
                ldsP[wave][j + hi * 8][lane15 + 16] = (__bf16)p1v[j];
            }
            asm volatile("s_wait_dscnt 0" ::: "memory");   // same-wave LDS RAW
            FragBF16 ap;
            ap.u[0] = ((const uint4*)&ldsP[wave][lane15][hi * 16])[0];
            ap.u[1] = ((const uint4*)&ldsP[wave][lane15][hi * 16 + 8])[0];

            // ---- O += P @ V : ping-pong V B-fragment registers
            FragBF16 f0, f1;
            auto loadV = [&](FragBF16& f, int nt) {
                const uint4* vp = (const uint4*)&ldsVT[buf][nt * 16 + lane15][hi * 16];
                f.u[0] = vp[0];
                f.u[1] = vp[1];
            };
            loadV(f0, 0);
            #pragma unroll
            for (int nt = 0; nt < 8; ++nt) {
                if (nt < 7) loadV((nt & 1) ? f0 : f1, nt + 1);
                o[nt] = wmma_bf16(ap.v, ((nt & 1) ? f1 : f0).v, o[nt]);
            }
        }
        __syncthreads();   // staging of buf^1 complete; buf free for reuse
    }

    // Epilogue: reference scales AFTER softmax by 1/sqrt(D).
    const float hs = 0.08838834764831845f;   // 1/sqrt(128)
    float inv[8];
    #pragma unroll
    for (int j = 0; j < 8; ++j) inv[j] = hs / l[j];
    #pragma unroll
    for (int nt = 0; nt < 8; ++nt) {
        const int col = nt * 16 + lane15;
        #pragma unroll
        for (int j = 0; j < 8; ++j) {
            const int row = qRow + j + hi * 8;
            out[(seqBase + row) * D_HEAD + col] = o[nt][j] * inv[j];
        }
    }
}

// -----------------------------------------------------------------------------
extern "C" void kernel_launch(void* const* d_in, const int* in_sizes, int n_in,
                              void* d_out, int out_size, void* d_ws, size_t ws_size,
                              hipStream_t stream)
{
    const float* x  = (const float*)d_in[0];
    const float* Wq = (const float*)d_in[1];
    const float* bq = (const float*)d_in[2];
    const float* Wk = (const float*)d_in[3];
    const float* bk = (const float*)d_in[4];
    const float* Wv = (const float*)d_in[5];
    const float* bv = (const float*)d_in[6];
    float* out = (float*)d_out;

    // Workspace: q/k/v in bf16, [B*S, D] each  -> 3 * 16384 * 128 * 2B = 12 MB
    const size_t elems = (size_t)B_SZ * S_LEN * D_HEAD;
    unsigned short* qws = (unsigned short*)d_ws;
    unsigned short* kws = qws + elems;
    unsigned short* vws = kws + elems;

    dim3 g1((B_SZ * S_LEN) / 64);      // 256 blocks, x read exactly once
    qkv_proj_fused<<<g1, 384, 0, stream>>>(x, Wq, bq, Wk, bk, Wv, bv, qws, kws, vws);

    dim3 g2(S_LEN / 64, B_SZ);         // 64 query rows per block
    flash_attn_kernel<<<g2, 128, 0, stream>>>(qws, kws, vws, out);
}
